// StartPointModel2_48567490183419
// MI455X (gfx1250) — compile-verified, matrix operand order checked by reading, not run on hardware
//
#include <hip/hip_runtime.h>

#define T_STEPS 512
#define BATCH   128
#define DIN     1024
#define H       256
#define G       1024   // 4*H
#define CHUNK   64
#define NVOCAB  3
#define HPAD    (H + 8)   // f16 rows: +16B so row stride = 132 dwords (mod 64 = 4)
#define FPAD    (H + 4)   // f32 rows: +16B so row stride = 260 dwords (mod 64 = 4)

typedef __attribute__((ext_vector_type(16))) _Float16 v16h;
typedef __attribute__((ext_vector_type(8)))  _Float16 v8h;
typedef __attribute__((ext_vector_type(8)))  float    v8f;

// ---------------------------------------------------------------------------
// WMMA helpers (CDNA5 wave32, 16x16x32 f16 -> f32)
// ---------------------------------------------------------------------------
static __device__ __forceinline__ v8f wmma_f16(v16h a, v16h b, v8f c) {
  // 8 args: (neg_a, A, neg_b, B, c_mod, C, reuse_a, reuse_b)
  return __builtin_amdgcn_wmma_f32_16x16x32_f16(false, a, false, b, (short)0, c,
                                                false, false);
}

// 16-bit A/B fragment: per-lane two contiguous 8-half chunks at k0 and k0+16
// (lane<16 -> k0 base 0, lane>=16 -> k0 base 8 within each 32-K block).
static __device__ __forceinline__ v16h ld_frag_h(const _Float16* p) {
  v8h lo = *(const v8h*)(p);
  v8h hi = *(const v8h*)(p + 16);
  return __builtin_shufflevector(lo, hi, 0, 1, 2, 3, 4, 5, 6, 7,
                                 8, 9, 10, 11, 12, 13, 14, 15);
}

// Same fragment but from an f32 source, converted to f16 on the fly.
static __device__ __forceinline__ v16h ld_frag_f32cvt(const float* p) {
  v16h a;
#pragma unroll
  for (int i = 0; i < 8; ++i) {
    a[i]     = (_Float16)p[i];
    a[i + 8] = (_Float16)p[i + 16];
  }
  return a;
}

static __device__ __forceinline__ float sigf(float x) {
  return 1.0f / (1.0f + __expf(-x));
}

// ---------------------------------------------------------------------------
// f32 -> f16 weight conversion
// ---------------------------------------------------------------------------
__global__ void cvt_f32_f16_kernel(const float* __restrict__ s,
                                   _Float16* __restrict__ d, int n) {
  int i = blockIdx.x * blockDim.x + threadIdx.x;
  int stride = gridDim.x * blockDim.x;
  for (; i < n; i += stride) d[i] = (_Float16)s[i];
}

// ---------------------------------------------------------------------------
// Phase 1: gates[t][b][g] = x[tt] @ W_ih.T + (b_ih + b_hh), one 64-step chunk,
// both directions.  One wave computes a 16(M) x 64(N) strip over K=1024.
// M rows = t*128 + b (tiles never straddle t since B is a multiple of 16).
// ---------------------------------------------------------------------------
__global__ __launch_bounds__(256)
void gates_gemm_kernel(const float* __restrict__ x,
                       const _Float16* __restrict__ wihF,
                       const _Float16* __restrict__ wihB,
                       const float* __restrict__ ebihF,
                       const float* __restrict__ ebhhF,
                       const float* __restrict__ ebihB,
                       const float* __restrict__ ebhhB,
                       float* __restrict__ gatesF,
                       float* __restrict__ gatesB,
                       int chunkStart) {
  const int lane  = threadIdx.x & 31;
  const int task  = blockIdx.x * 8 + (threadIdx.x >> 5);  // 16384 wave tasks
  const int dir   = task >> 13;                           // 8192 per direction
  const int rem   = task & 8191;
  const int mtile = rem & 511;                            // 512 M-tiles of 16
  const int ntile = rem >> 9;                             // 16 N-strips of 64

  const int r0 = mtile * 16;          // row in [CHUNK*BATCH)
  const int t  = r0 >> 7;             // local chunk step
  const int b0 = r0 & 127;            // batch base
  const int tt = (dir == 0) ? (chunkStart + t)
                            : (T_STEPS - 1 - (chunkStart + t));

  const _Float16* w  = (dir == 0) ? wihF : wihB;
  const float* bi    = (dir == 0) ? ebihF : ebihB;
  const float* bh    = (dir == 0) ? ebhhF : ebhhB;
  float* gout        = (dir == 0) ? gatesF : gatesB;

  const int am  = lane & 15;          // A row within tile / D column within tile
  const int hi8 = (lane >> 4) * 8;    // K-chunk base (A/B) and row base (C/D)

  const float* xrow = x + ((size_t)(tt * BATCH + b0 + am)) * DIN;

  v8f acc[4] = {v8f{}, v8f{}, v8f{}, v8f{}};

#pragma unroll 1
  for (int kb = 0; kb < DIN / 32; ++kb) {
    const int k0 = kb * 32 + hi8;
    v16h a = ld_frag_f32cvt(xrow + k0);
#pragma unroll
    for (int j = 0; j < 4; ++j) {
      const int n = ntile * 64 + j * 16 + am;
      v16h bf = ld_frag_h(w + (size_t)n * DIN + k0);
      acc[j] = wmma_f16(a, bf, acc[j]);
    }
  }

  // epilogue: add folded bias and scatter-store (lane = column, VGPR = row)
#pragma unroll
  for (int j = 0; j < 4; ++j) {
    const int n = ntile * 64 + j * 16 + am;
    const float bias = bi[n] + bh[n];
    float* o = gout + ((size_t)(t * BATCH + b0 + hi8)) * G + n;
#pragma unroll
    for (int e = 0; e < 8; ++e) o[(size_t)e * G] = acc[j][e] + bias;
  }
}

// ---------------------------------------------------------------------------
// Phase 2: encoder recurrence for one 64-step chunk.
// 16 blocks = {dir} x {8 batch tiles of 16}; 512 threads = 16 waves.
// Wave wv owns H-columns [16*wv, 16*wv+16) for ALL FOUR gates, so i/f/g/o
// land in matching lane/VGPR slots and the LSTM update is register-only.
// h lives in LDS (f16, the A operand for the next step); c lives in VGPRs.
// ---------------------------------------------------------------------------
__global__ __launch_bounds__(512)
void encoder_chunk_kernel(const float* __restrict__ gatesF,
                          const float* __restrict__ gatesB,
                          const _Float16* __restrict__ whhF,
                          const _Float16* __restrict__ whhB,
                          float* __restrict__ hstate,
                          float* __restrict__ cstate,
                          int chunkStart) {
  __shared__ _Float16 hsm[16][HPAD];

  const int lane  = threadIdx.x & 31;
  const int wv    = threadIdx.x >> 5;       // 0..15 column tile
  const int dir   = blockIdx.x >> 3;
  const int btile = blockIdx.x & 7;
  const int b0    = btile * 16;

  const float* gates  = (dir == 0) ? gatesF : gatesB;
  const _Float16* whh = (dir == 0) ? whhF : whhB;
  float* hst = hstate + (size_t)dir * BATCH * H;
  float* cst = cstate + (size_t)dir * BATCH * H;

  const int am  = lane & 15;
  const int hi8 = (lane >> 4) * 8;          // row base of the C/D fragment
  const int col = wv * 16 + am;             // owned H column

  float creg[8], hreg[8];
  if (chunkStart == 0) {
#pragma unroll
    for (int e = 0; e < 8; ++e) { creg[e] = 0.0f; hreg[e] = 0.0f; }
  } else {
#pragma unroll
    for (int e = 0; e < 8; ++e) {
      creg[e] = cst[(size_t)(b0 + hi8 + e) * H + col];
      hreg[e] = hst[(size_t)(b0 + hi8 + e) * H + col];
    }
  }
#pragma unroll
  for (int e = 0; e < 8; ++e) hsm[hi8 + e][col] = (_Float16)hreg[e];
  __syncthreads();

#pragma unroll 1
  for (int t = 0; t < CHUNK; ++t) {
    v8f acc[4];
#pragma unroll
    for (int q = 0; q < 4; ++q) {
      const float* gp =
          gates + ((size_t)(t * BATCH + b0 + hi8)) * G + q * H + col;
#pragma unroll
      for (int e = 0; e < 8; ++e) acc[q][e] = gp[(size_t)e * G];
    }
#pragma unroll 1
    for (int kb = 0; kb < H / 32; ++kb) {
      const int k0 = kb * 32 + hi8;
      v16h a = ld_frag_h(&hsm[am][0] + k0);
#pragma unroll
      for (int q = 0; q < 4; ++q) {
        v16h bf = ld_frag_h(whh + (size_t)(q * H + col) * H + k0);
        acc[q] = wmma_f16(a, bf, acc[q]);
      }
    }
    __syncthreads();  // all waves done reading hsm for this step
#pragma unroll
    for (int e = 0; e < 8; ++e) {
      const float gi = acc[0][e], gf = acc[1][e], gg = acc[2][e], go = acc[3][e];
      const float c = sigf(gf) * creg[e] + sigf(gi) * tanhf(gg);
      const float h = sigf(go) * tanhf(c);
      creg[e] = c;
      hreg[e] = h;
      hsm[hi8 + e][col] = (_Float16)h;
    }
    __syncthreads();  // hsm ready for next step
  }

#pragma unroll
  for (int e = 0; e < 8; ++e) {
    cst[(size_t)(b0 + hi8 + e) * H + col] = creg[e];
    hst[(size_t)(b0 + hi8 + e) * H + col] = hreg[e];
  }
}

// ---------------------------------------------------------------------------
// Phase 3: 2-layer autoregressive decoder, 64 steps, feedback out = h1.
// 8 blocks (batch tiles of 16), 512 threads = 16 waves, same tiling scheme.
// layer0 init (h,c) = encoder fwd final; layer1 init = encoder bwd final.
// ---------------------------------------------------------------------------
__global__ __launch_bounds__(512)
void decoder_kernel(const float* __restrict__ hstate,
                    const float* __restrict__ cstate,
                    const _Float16* __restrict__ wih0,
                    const _Float16* __restrict__ whh0,
                    const _Float16* __restrict__ wih1,
                    const _Float16* __restrict__ whh1,
                    const float* __restrict__ bih0,
                    const float* __restrict__ bhh0,
                    const float* __restrict__ bih1,
                    const float* __restrict__ bhh1,
                    const float* __restrict__ lw,
                    const float* __restrict__ lb,
                    float* __restrict__ out) {
  __shared__ _Float16 inp_sm[16][HPAD];  // decoder input (= previous h1), f16
  __shared__ _Float16 h0_sm[16][HPAD];   // layer0 hidden, f16
  __shared__ _Float16 h1_sm[16][HPAD];   // layer1 hidden, f16
  __shared__ float    h1f[16][FPAD];     // layer1 hidden, f32 (for projection)

  const int lane = threadIdx.x & 31;
  const int wv   = threadIdx.x >> 5;
  const int b0   = blockIdx.x * 16;

  const int am  = lane & 15;
  const int hi8 = (lane >> 4) * 8;
  const int col = wv * 16 + am;

  float c0[8], c1[8];
#pragma unroll
  for (int e = 0; e < 8; ++e) {
    const int b = b0 + hi8 + e;
    c0[e] = cstate[(size_t)b * H + col];                       // cf (fwd)
    c1[e] = cstate[(size_t)BATCH * H + (size_t)b * H + col];   // cb (bwd)
    h0_sm[hi8 + e][col]  = (_Float16)hstate[(size_t)b * H + col];
    h1_sm[hi8 + e][col]  = (_Float16)hstate[(size_t)BATCH * H + (size_t)b * H + col];
    inp_sm[hi8 + e][col] = (_Float16)0.0f;
  }
  // per-column gate biases are loop-invariant: load once
  float bias0[4], bias1[4];
#pragma unroll
  for (int q = 0; q < 4; ++q) {
    bias0[q] = bih0[q * H + col] + bhh0[q * H + col];
    bias1[q] = bih1[q * H + col] + bhh1[q * H + col];
  }
  __syncthreads();

#pragma unroll 1
  for (int t = 0; t < CHUNK; ++t) {
    // ---- layer 0: gates = inp@wih0.T + h0@whh0.T + b ----
    v8f acc[4];
#pragma unroll
    for (int q = 0; q < 4; ++q) {
#pragma unroll
      for (int e = 0; e < 8; ++e) acc[q][e] = bias0[q];
    }
#pragma unroll 1
    for (int kb = 0; kb < H / 32; ++kb) {
      const int k0 = kb * 32 + hi8;
      v16h ai = ld_frag_h(&inp_sm[am][0] + k0);
#pragma unroll
      for (int q = 0; q < 4; ++q) {
        v16h bf = ld_frag_h(wih0 + (size_t)(q * H + col) * H + k0);
        acc[q] = wmma_f16(ai, bf, acc[q]);
      }
      v16h ah = ld_frag_h(&h0_sm[am][0] + k0);
#pragma unroll
      for (int q = 0; q < 4; ++q) {
        v16h bf = ld_frag_h(whh0 + (size_t)(q * H + col) * H + k0);
        acc[q] = wmma_f16(ah, bf, acc[q]);
      }
    }
    __syncthreads();
#pragma unroll
    for (int e = 0; e < 8; ++e) {
      const float gi = acc[0][e], gf = acc[1][e], gg = acc[2][e], go = acc[3][e];
      const float c = sigf(gf) * c0[e] + sigf(gi) * tanhf(gg);
      const float h = sigf(go) * tanhf(c);
      c0[e] = c;
      h0_sm[hi8 + e][col] = (_Float16)h;
    }
    __syncthreads();

    // ---- layer 1: gates = h0@wih1.T + h1@whh1.T + b ----
#pragma unroll
    for (int q = 0; q < 4; ++q) {
#pragma unroll
      for (int e = 0; e < 8; ++e) acc[q][e] = bias1[q];
    }
#pragma unroll 1
    for (int kb = 0; kb < H / 32; ++kb) {
      const int k0 = kb * 32 + hi8;
      v16h ai = ld_frag_h(&h0_sm[am][0] + k0);
#pragma unroll
      for (int q = 0; q < 4; ++q) {
        v16h bf = ld_frag_h(wih1 + (size_t)(q * H + col) * H + k0);
        acc[q] = wmma_f16(ai, bf, acc[q]);
      }
      v16h ah = ld_frag_h(&h1_sm[am][0] + k0);
#pragma unroll
      for (int q = 0; q < 4; ++q) {
        v16h bf = ld_frag_h(whh1 + (size_t)(q * H + col) * H + k0);
        acc[q] = wmma_f16(ah, bf, acc[q]);
      }
    }
    __syncthreads();
#pragma unroll
    for (int e = 0; e < 8; ++e) {
      const float gi = acc[0][e], gf = acc[1][e], gg = acc[2][e], go = acc[3][e];
      const float c = sigf(gf) * c1[e] + sigf(gi) * tanhf(gg);
      const float h = sigf(go) * tanhf(c);
      c1[e] = c;
      h1_sm[hi8 + e][col]  = (_Float16)h;  // hidden for next step
      inp_sm[hi8 + e][col] = (_Float16)h;  // feedback input for next step
      h1f[hi8 + e][col]    = h;
    }
    __syncthreads();

    // ---- vocab projection: out[t,b,:] = h1 @ lw.T + lb ----
    if (threadIdx.x < 16 * NVOCAB) {
      const int b = threadIdx.x / NVOCAB;
      const int v = threadIdx.x % NVOCAB;
      float s = lb[v];
#pragma unroll 4
      for (int j = 0; j < H; ++j) s += h1f[b][j] * lw[v * H + j];
      out[((size_t)t * BATCH + b0 + b) * NVOCAB + v] = s;
    }
    __syncthreads();
  }
}

// ---------------------------------------------------------------------------
// Host entry
// ---------------------------------------------------------------------------
extern "C" void kernel_launch(void* const* d_in, const int* in_sizes, int n_in,
                              void* d_out, int out_size, void* d_ws,
                              size_t ws_size, hipStream_t stream) {
  (void)in_sizes; (void)n_in; (void)out_size; (void)ws_size;

  const float* x       = (const float*)d_in[0];
  const float* ew_ih_f = (const float*)d_in[1];
  const float* ew_hh_f = (const float*)d_in[2];
  const float* eb_ih_f = (const float*)d_in[3];
  const float* eb_hh_f = (const float*)d_in[4];
  const float* ew_ih_b = (const float*)d_in[5];
  const float* ew_hh_b = (const float*)d_in[6];
  const float* eb_ih_b = (const float*)d_in[7];
  const float* eb_hh_b = (const float*)d_in[8];
  const float* dw_ih0  = (const float*)d_in[9];
  const float* dw_hh0  = (const float*)d_in[10];
  const float* db_ih0  = (const float*)d_in[11];
  const float* db_hh0  = (const float*)d_in[12];
  const float* dw_ih1  = (const float*)d_in[13];
  const float* dw_hh1  = (const float*)d_in[14];
  const float* db_ih1  = (const float*)d_in[15];
  const float* db_hh1  = (const float*)d_in[16];
  const float* lw      = (const float*)d_in[17];
  const float* lb      = (const float*)d_in[18];
  float* out = (float*)d_out;

  char* ws = (char*)d_ws;
  size_t off = 0;
  auto alloc = [&](size_t bytes) -> void* {
    void* p = ws + off;
    off += (bytes + 255) & ~(size_t)255;
    return p;
  };

  float* gatesF = (float*)alloc((size_t)CHUNK * BATCH * G * sizeof(float));
  float* gatesB = (float*)alloc((size_t)CHUNK * BATCH * G * sizeof(float));
  float* hstate = (float*)alloc((size_t)2 * BATCH * H * sizeof(float));
  float* cstate = (float*)alloc((size_t)2 * BATCH * H * sizeof(float));
  _Float16* wihFh = (_Float16*)alloc((size_t)G * DIN * 2);
  _Float16* wihBh = (_Float16*)alloc((size_t)G * DIN * 2);
  _Float16* whhFh = (_Float16*)alloc((size_t)G * H * 2);
  _Float16* whhBh = (_Float16*)alloc((size_t)G * H * 2);
  _Float16* wih0h = (_Float16*)alloc((size_t)G * H * 2);
  _Float16* whh0h = (_Float16*)alloc((size_t)G * H * 2);
  _Float16* wih1h = (_Float16*)alloc((size_t)G * H * 2);
  _Float16* whh1h = (_Float16*)alloc((size_t)G * H * 2);

  auto cvt = [&](const float* s, _Float16* d, int n) {
    int grid = (n + 255) / 256;
    if (grid > 2048) grid = 2048;
    cvt_f32_f16_kernel<<<grid, 256, 0, stream>>>(s, d, n);
  };
  cvt(ew_ih_f, wihFh, G * DIN);
  cvt(ew_ih_b, wihBh, G * DIN);
  cvt(ew_hh_f, whhFh, G * H);
  cvt(ew_hh_b, whhBh, G * H);
  cvt(dw_ih0, wih0h, G * H);
  cvt(dw_hh0, whh0h, G * H);
  cvt(dw_ih1, wih1h, G * H);
  cvt(dw_hh1, whh1h, G * H);

  // Chunked: big WMMA GEMM for x@W_ih.T (both dirs), then batch-parallel
  // recurrence over those 64 steps.  Stream order provides the dependency.
  for (int cs = 0; cs < T_STEPS; cs += CHUNK) {
    gates_gemm_kernel<<<2048, 256, 0, stream>>>(
        x, wihFh, wihBh, eb_ih_f, eb_hh_f, eb_ih_b, eb_hh_b, gatesF, gatesB,
        cs);
    encoder_chunk_kernel<<<16, 512, 0, stream>>>(gatesF, gatesB, whhFh, whhBh,
                                                 hstate, cstate, cs);
  }

  decoder_kernel<<<8, 512, 0, stream>>>(hstate, cstate, wih0h, whh0h, wih1h,
                                        whh1h, db_ih0, db_hh0, db_ih1, db_hh1,
                                        lw, lb, out);
}
